// RNN_4904852652923
// MI455X (gfx1250) — compile-verified, matrix operand order checked by reading, not run on hardware
//
#include <hip/hip_runtime.h>
#include <hip/hip_bf16.h>

#define BATCH 256
#define SEQ   2048
#define DIM   64
#define HID   100
#define HP    128      // hidden padded to 8 x 16 for WMMA tiling
#define TCHUNK 64      // timesteps per phase-1 block

typedef __attribute__((ext_vector_type(2)))  float     v2f;
typedef __attribute__((ext_vector_type(4)))  float     v4f;
typedef __attribute__((ext_vector_type(8)))  float     v8f;
typedef __attribute__((ext_vector_type(8)))  _Float16  v8h;
typedef __attribute__((ext_vector_type(16))) _Float16  v16h;
typedef __attribute__((ext_vector_type(4)))  unsigned int v4u;
typedef __attribute__((ext_vector_type(8)))  unsigned int v8u;

// ---------------------------------------------------------------------------
// TDM: async DMA of one 16-row x 64-col f32 x-tile (row stride SEQ*DIM elems)
// from global into LDS. Descriptor per CDNA5 ISA.
// 2D tensor -> D# groups 2/3 are NULL (2-operand form). Tracked by TENSORcnt.
// ---------------------------------------------------------------------------
__device__ __forceinline__ void tdm_load_x_tile(unsigned int lds_byte_addr,
                                                const float* gptr)
{
    unsigned long long ga = (unsigned long long)(uintptr_t)gptr;

    v4u g0;
    g0[0] = 1u;                                   // count=1, user D#, no gather
    g0[1] = lds_byte_addr;                        // lds_addr [63:32]
    g0[2] = (unsigned int)ga;                     // global_addr [95:64]
    g0[3] = (unsigned int)((ga >> 32) & 0x1FFFFFFu) // global_addr [120:96]
          | 0x80000000u;                          // type=2 ("image") [127:126]

    v8u g1;
    g1[0] = 2u << 16;                             // data_size = 2 -> 4 bytes
    g1[1] = (unsigned int)DIM << 16;              // tensor_dim0 = 64  [79:48]
    g1[2] = 16u << 16;                            // tensor_dim1 = 16  [111:80]
    g1[3] = (unsigned int)DIM << 16;              // tile_dim0  = 64  [127:112]
    g1[4] = 16u;                                  // tile_dim1  = 16  [143:128]
    g1[5] = (unsigned int)(SEQ * DIM);            // tensor_dim0_stride [207:160]
    g1[6] = 0u;
    g1[7] = 0u;

    asm volatile("tensor_load_to_lds %0, %1" :: "s"(g0), "s"(g1) : "memory");
}

// ---------------------------------------------------------------------------
// Phase 1: pre[t,b,n] = x[b,t,:] @ Wx[:,n] + bias[n], stored into d_out.
// f32 WMMA 16x16x4. Block = 224 threads = 7 waves, one wave per 16-col n-tile
// (7*16 = 112 >= 100). Block handles one 16-row batch tile for TCHUNK steps.
// x-tiles arrive via double-buffered TDM DMA overlapped with WMMA compute.
// ---------------------------------------------------------------------------
__global__ __launch_bounds__(224)
void rnn_phase1_xw(const float* __restrict__ x, const float* __restrict__ W,
                   const float* __restrict__ bias, float* __restrict__ pre)
{
    __shared__ __align__(16) float xs[2][16 * DIM];   // 2 x 4 KB ping-pong

    const int wave  = threadIdx.x >> 5;            // 0..6 -> n-tile
    const int lane  = threadIdx.x & 31;
    const int btile = blockIdx.x & 15;             // 16 batch tiles
    const int tch   = blockIdx.x >> 4;             // 32 time chunks
    const int bbase = btile * 16;
    const int tbase = tch * TCHUNK;
    const int nn    = wave * 16 + (lane & 15);     // output column
    const int khalf = (lane >> 4) ? 2 : 0;         // f32 A/B: hi lanes hold K+2,K+3
    const bool nv   = (nn < HID);

    // Preload B fragments of Wx: 16 fragments of K=4; lane holds column nn,
    // two consecutive K values per fragment (v2f).
    v2f bfrag[16];
    #pragma unroll
    for (int j = 0; j < 16; ++j) {
        int k = 4 * j + khalf;
        bfrag[j].x = nv ? W[k * HID + nn]       : 0.f;
        bfrag[j].y = nv ? W[(k + 1) * HID + nn] : 0.f;
    }
    const float bias_n = nv ? bias[nn] : 0.f;

    // Kick off DMA of the first x-tile (wave 0 only: TDM ignores EXEC, so
    // gate at wave granularity to issue exactly one descriptor).
    if (threadIdx.x < 32) {
        tdm_load_x_tile((unsigned int)(uintptr_t)(void*)&xs[0][0],
                        x + ((size_t)bbase * SEQ + tbase) * DIM);
        __builtin_amdgcn_s_wait_tensorcnt(0);
    }
    __syncthreads();

    int cur = 0;
    for (int tt = 0; tt < TCHUNK; ++tt) {
        const int t = tbase + tt;

        // async prefetch of the next tile into the other buffer
        if (threadIdx.x < 32 && (tt + 1) < TCHUNK) {
            tdm_load_x_tile((unsigned int)(uintptr_t)(void*)&xs[cur ^ 1][0],
                            x + ((size_t)bbase * SEQ + (t + 1)) * DIM);
        }

        v8f c;
        #pragma unroll
        for (int v = 0; v < 8; ++v) c[v] = bias_n;

        const float* xt = &xs[cur][0];
        const int arow = lane & 15;                // A: lane holds row M=arow
        #pragma unroll
        for (int j = 0; j < 16; ++j) {
            v2f a = *(const v2f*)&xt[arow * DIM + 4 * j + khalf];
            c = __builtin_amdgcn_wmma_f32_16x16x4_f32(
                    false, a, false, bfrag[j], (short)0, c, false, false);
        }

        // C layout: element (M = v + 8*(lane>=16), N = lane&15)
        if (nv) {
            const int mhi = (lane >> 4) ? 8 : 0;
            size_t base = (size_t)t * BATCH * HID + (size_t)(bbase + mhi) * HID + nn;
            #pragma unroll
            for (int v = 0; v < 8; ++v)
                pre[base + (size_t)v * HID] = c[v];
        }

        // DMA for the next tile must land before anyone reads it
        if (threadIdx.x < 32)
            __builtin_amdgcn_s_wait_tensorcnt(0);
        __syncthreads();
        cur ^= 1;
    }
}

// ---------------------------------------------------------------------------
// Phase 2: serial scan. 16 blocks x 8 waves (256 thr). Block owns 16 batch
// rows; wave w owns hidden columns [16w, 16w+16). Wh held in registers as
// f16 B-fragments (K padded to 128 with zeros => padded h columns stay 0).
// Per step: prefetch pre[t+1], 4x v_wmma_f32_16x16x32_f16, tanh, store h_t
// to d_out, f16 h into ping-pong LDS, one barrier.
// ---------------------------------------------------------------------------
__global__ __launch_bounds__(256)
void rnn_phase2_scan(const float* __restrict__ W, float* __restrict__ out)
{
    __shared__ __align__(16) _Float16 hbuf[2][16 * HP];   // 2 x 4 KB ping-pong

    const int wave  = threadIdx.x >> 5;
    const int lane  = threadIdx.x & 31;
    const int bbase = blockIdx.x * 16;
    const int nloc  = wave * 16 + (lane & 15);  // padded hidden col 0..127
    const bool nv   = (nloc < HID);
    const int  mhi  = (lane >> 4) ? 8 : 0;

    // zero initial h (h0 = 0)
    for (int i = threadIdx.x; i < 16 * HP; i += 256)
        hbuf[0][i] = (_Float16)0;

    // Preload Wh as f16 B fragments: 4 fragments of K=32.
    // 16-bit B layout (mirror of A): lo lanes: halves 0..7 -> K+0..7,
    // halves 8..15 -> K+16..23; hi lanes offset by +8.
    const float* Wh = W + DIM * HID;
    v16h bfrag[4];
    #pragma unroll
    for (int f = 0; f < 4; ++f) {
        #pragma unroll
        for (int i = 0; i < 16; ++i) {
            int koff = (i & 7) + ((i >> 3) << 4) + ((lane >> 4) ? 8 : 0);
            int k = 32 * f + koff;
            float val = (nv && k < HID) ? Wh[k * HID + nloc] : 0.f;
            bfrag[f][i] = (_Float16)val;
        }
    }
    __syncthreads();

    // prefetch pre for t=0 (phase 1 wrote it into `out`)
    v8f pre_cur;
    {
        size_t base = (size_t)(bbase + mhi) * HID + nloc;
        #pragma unroll
        for (int v = 0; v < 8; ++v)
            pre_cur[v] = nv ? out[base + (size_t)v * HID] : 0.f;
    }

    int cur = 0;
    for (int t = 0; t < SEQ; ++t) {
        // software-pipelined prefetch of next step's pre-activations
        v8f pre_next;
        if (t + 1 < SEQ) {
            size_t base = (size_t)(t + 1) * BATCH * HID
                        + (size_t)(bbase + mhi) * HID + nloc;
            #pragma unroll
            for (int v = 0; v < 8; ++v)
                pre_next[v] = nv ? out[base + (size_t)v * HID] : 0.f;
        } else {
            #pragma unroll
            for (int v = 0; v < 8; ++v) pre_next[v] = 0.f;
        }

        // c = pre[t] + h_{t-1} @ Wh
        v8f c = pre_cur;
        const _Float16* h = hbuf[cur];
        const int arow  = lane & 15;
        const int abase = (lane >> 4) ? 8 : 0;
        #pragma unroll
        for (int f = 0; f < 4; ++f) {
            // A 16x32 f16 layout: lane = row; lo lanes K 0..7 & 16..23,
            // hi lanes K 8..15 & 24..31 => two 16B LDS reads per fragment.
            v8h lo = *(const v8h*)&h[arow * HP + 32 * f + abase];
            v8h hi = *(const v8h*)&h[arow * HP + 32 * f + abase + 16];
            v16h a;
            #pragma unroll
            for (int i = 0; i < 8; ++i) { a[i] = lo[i]; a[i + 8] = hi[i]; }
            c = __builtin_amdgcn_wmma_f32_16x16x32_f16(
                    false, a, false, bfrag[f], (short)0, c, false, false);
        }

        // activation (padded cols forced to 0 so they never pollute the scan)
        v8f hnew;
        #pragma unroll
        for (int v = 0; v < 8; ++v) hnew[v] = nv ? tanhf(c[v]) : 0.f;

        // write h_t over pre[t] in d_out
        if (nv) {
            size_t base = (size_t)t * BATCH * HID
                        + (size_t)(bbase + mhi) * HID + nloc;
            #pragma unroll
            for (int v = 0; v < 8; ++v)
                out[base + (size_t)v * HID] = hnew[v];
        }

        // publish f16 h_t into the other LDS buffer
        _Float16* hn = hbuf[cur ^ 1];
        #pragma unroll
        for (int v = 0; v < 8; ++v)
            hn[(mhi + v) * HP + nloc] = (_Float16)hnew[v];

        __syncthreads();   // one barrier per step (ping-pong removes the 2nd)
        cur ^= 1;
        pre_cur = pre_next;
    }
}

// ---------------------------------------------------------------------------
extern "C" void kernel_launch(void* const* d_in, const int* in_sizes, int n_in,
                              void* d_out, int out_size, void* d_ws, size_t ws_size,
                              hipStream_t stream) {
    const float* x  = (const float*)d_in[0];   // (256, 2048, 64)
    const float* W  = (const float*)d_in[1];   // (164, 100)
    const float* b  = (const float*)d_in[2];   // (100,)
    float* out      = (float*)d_out;           // (2048, 256, 100)

    dim3 g1(16 * (SEQ / TCHUNK));              // 16 btiles x 32 t-chunks = 512
    rnn_phase1_xw<<<g1, 224, 0, stream>>>(x, W, b, out);
    rnn_phase2_scan<<<16, 256, 0, stream>>>(W, out);
}
